// TLSTM_80831284510899
// MI455X (gfx1250) — matrix operand order repaired and checked by reading, not compiled, not run on hardware
//
#include <hip/hip_runtime.h>
#include <hip/hip_bf16.h>

// ---------------------------------------------------------------------------
// TLSTM for MI455X (gfx1250): single persistent bf16-WMMA kernel.
// B=256, S=512, D=512, H=512, FC=64, O=2
// Each wave owns a 16x16 state patch and computes ALL five GEMM accumulators
// for it (i/f/o/c~ over K=1024 of [x_t|h], decomp over K=512 of c), then does
// the gate math in registers. One grid barrier per step, ping-pong bf16 state.
// ---------------------------------------------------------------------------

typedef __attribute__((ext_vector_type(16))) __bf16 v16bf;
typedef __attribute__((ext_vector_type(8)))  float  v8f;
typedef __attribute__((ext_vector_type(4)))  int    v4i;
typedef __attribute__((ext_vector_type(4)))  float  v4f;

#define BATCH 256
#define SEQ   512
#define DIN   512
#define HID   512
#define ECONST 2.7183f
#define NBLK  64          // persistent blocks (512 waves total)

#define WMMA_BF16(A, Bv, C) \
    __builtin_amdgcn_wmma_f32_16x16x32_bf16(false, (A), false, (Bv), (short)0, (C), false, false)

// ---------------------------------------------------------------------------
// Fragment loaders (CDNA5 ISA 7.12.2 layouts, wave32)
// ---------------------------------------------------------------------------

// A-matrix 16x32 bf16 fragment from row-major bf16.
__device__ inline v16bf a_frag_bf16(const __bf16* A, long lda,
                                    int m0, int k0, int lane) {
    int r = lane & 15, hf = lane >> 4;
    const __bf16* p = A + (long)(m0 + r) * lda + k0 + hf * 8;
    v16bf f;
    ((v4i*)&f)[0] = *(const v4i*)p;          // K offsets 0..7
    ((v4i*)&f)[1] = *(const v4i*)(p + 16);   // K offsets 16..23
    return f;
}

// Same A fragment from row-major f32 (converted to bf16 in-flight).
__device__ inline v16bf a_frag_f32(const float* A, long lda,
                                   int m0, int k0, int lane) {
    int r = lane & 15, hf = lane >> 4;
    const float* p = A + (long)(m0 + r) * lda + k0 + hf * 8;
    v4f q0 = ((const v4f*)p)[0];
    v4f q1 = ((const v4f*)p)[1];
    v4f q2 = ((const v4f*)(p + 16))[0];
    v4f q3 = ((const v4f*)(p + 16))[1];
    v16bf f;
#pragma unroll
    for (int i = 0; i < 4; ++i) {
        f[i]      = (__bf16)q0[i];
        f[4 + i]  = (__bf16)q1[i];
        f[8 + i]  = (__bf16)q2[i];
        f[12 + i] = (__bf16)q3[i];
    }
    return f;
}

// B-matrix 32x16 bf16 fragment from pre-packed buffer:
// [ktile][ntile][lane][16 bf16] -> one 32B contiguous chunk per lane.
__device__ inline v16bf b_frag(const __bf16* Bp, int ntiles,
                               int kt, int nt, int lane) {
    const __bf16* p = Bp + (((size_t)kt * ntiles + nt) * 32 + lane) * 16;
    v16bf f;
    ((v4i*)&f)[0] = ((const v4i*)p)[0];
    ((v4i*)&f)[1] = ((const v4i*)p)[1];
    return f;
}

// ---------------------------------------------------------------------------
// Device-wide barrier (sense-reversal via generation counter).
// ---------------------------------------------------------------------------
__device__ inline void grid_sync(unsigned* cnt, unsigned* gen) {
    __syncthreads();
    if (threadIdx.x == 0) {
        __threadfence();
        unsigned g = __hip_atomic_load(gen, __ATOMIC_RELAXED,
                                       __HIP_MEMORY_SCOPE_AGENT);
        unsigned prev = __hip_atomic_fetch_add(cnt, 1u, __ATOMIC_ACQ_REL,
                                               __HIP_MEMORY_SCOPE_AGENT);
        if (prev == NBLK - 1) {
            __hip_atomic_store(cnt, 0u, __ATOMIC_RELAXED,
                               __HIP_MEMORY_SCOPE_AGENT);
            __hip_atomic_fetch_add(gen, 1u, __ATOMIC_RELEASE,
                                   __HIP_MEMORY_SCOPE_AGENT);
        } else {
            while (__hip_atomic_load(gen, __ATOMIC_ACQUIRE,
                                     __HIP_MEMORY_SCOPE_AGENT) == g)
                __builtin_amdgcn_s_sleep(2);
        }
        __threadfence();
    }
    __syncthreads();
}

// ---------------------------------------------------------------------------
// Weight packing: f32 row-major -> bf16 WMMA B-fragment layout.
// Bp0: [1024 x 2048] rows 0..511=[Wi|Wf|Wog|Wc], rows 512..1023=[Ui|Uf|Uog|Uc]
// Bp1: [ 512 x  512] W_decomp
// Fragment element e of lane l in tile (kt,nt):
//   K = kt*32 + (l>>4)*8 + e + (e>=8 ? 8 : 0) ; N = nt*16 + (l&15)
// ---------------------------------------------------------------------------
__global__ void pack_weights_kernel(
    const float* __restrict__ Wi, const float* __restrict__ Wf,
    const float* __restrict__ Wog, const float* __restrict__ Wc,
    const float* __restrict__ Ui, const float* __restrict__ Uf,
    const float* __restrict__ Uog, const float* __restrict__ Uc,
    const float* __restrict__ Wd,
    __bf16* __restrict__ Bp0, __bf16* __restrict__ Bp1) {
    int t = blockIdx.x * blockDim.x + threadIdx.x;
    int lane = t & 31;
    int tile = t >> 5;
    const int NT0 = 2048 / 16;          // 128
    const int KT0 = 1024 / 32;          // 32
    if (tile < KT0 * NT0) {
        int nt = tile % NT0;
        __bf16* dst = Bp0 + ((size_t)tile * 32 + lane) * 16;
        int n = nt * 16 + (lane & 15);
        int kbase = (tile / NT0) * 32 + (lane >> 4) * 8;
        int g = n >> 9, nn = n & 511;
        const float* Wg = (g == 0) ? Wi : (g == 1) ? Wf : (g == 2) ? Wog : Wc;
        const float* Ug = (g == 0) ? Ui : (g == 1) ? Uf : (g == 2) ? Uog : Uc;
#pragma unroll
        for (int e = 0; e < 16; ++e) {
            int k = kbase + e + (e >= 8 ? 8 : 0);
            float v = (k < 512) ? Wg[(size_t)k * 512 + nn]
                                : Ug[(size_t)(k - 512) * 512 + nn];
            dst[e] = (__bf16)v;
        }
    } else {
        tile -= KT0 * NT0;
        const int NT1 = 512 / 16;       // 32 ; KT1 = 16
        if (tile < 16 * NT1) {
            int nt = tile % NT1;
            __bf16* dst = Bp1 + ((size_t)tile * 32 + lane) * 16;
            int n = nt * 16 + (lane & 15);
            int kbase = (tile / NT1) * 32 + (lane >> 4) * 8;
#pragma unroll
            for (int e = 0; e < 16; ++e)
                dst[e] = (__bf16)Wd[(size_t)(kbase + e + (e >= 8 ? 8 : 0)) * 512 + n];
        }
    }
}

__global__ void zero_barrier_kernel(unsigned* cnt, unsigned* gen) {
    *cnt = 0u;
    *gen = 0u;
}

// ---------------------------------------------------------------------------
// Persistent recurrence kernel: 64 blocks x 256 threads = 512 waves.
// wave -> (mt = wave & 15, g = wave >> 4): rows mt*16..+15, cols g*16..+15.
// ---------------------------------------------------------------------------
__global__ __launch_bounds__(256) void tlstm_persistent_kernel(
    const float* __restrict__ x, const float* __restrict__ time,
    const __bf16* __restrict__ Bp0, const __bf16* __restrict__ Bp1,
    const float* __restrict__ bi, const float* __restrict__ bfv,
    const float* __restrict__ bog, const float* __restrict__ bc,
    const float* __restrict__ bd,
    float* c32, float* h32,
    __bf16* hbf0, __bf16* cbf0, __bf16* hbf1, __bf16* cbf1,
    unsigned* barcnt, unsigned* bargen) {
    int lane = threadIdx.x & 31;
    int wave = blockIdx.x * 8 + (threadIdx.x >> 5);   // 0..511
    int g  = wave >> 4;                                // column group 0..31
    int mt = wave & 15;                                // row tile 0..15
    int m0 = mt << 4;
    int j0 = g << 4;
    int r = lane & 15, hf = lane >> 4;
    int col  = j0 + r;                                 // this lane's column
    int row0 = m0 + hf * 8;                            // first of 8 owned rows

    // per-lane biases for its column (constant across steps)
    float Bi = bi[col], Bf = bfv[col], Bo = bog[col],
          Bc = bc[col], Bd = bd[col];

    // zero the owned state patch (each (row,col) covered exactly once)
#pragma unroll
    for (int v = 0; v < 8; ++v) {
        size_t o = (size_t)(row0 + v) * HID + col;
        c32[o] = 0.f;
        hbf0[o] = (__bf16)0.f;
        cbf0[o] = (__bf16)0.f;
    }
    grid_sync(barcnt, bargen);

    __bf16 *hsrc = hbf0, *csrc = cbf0, *hdst = hbf1, *cdst = cbf1;

    for (int s = 0; s < SEQ; ++s) {
        v8f acc_i = {}, acc_f = {}, acc_o = {}, acc_c = {}, acc_d = {};

        const float* xbase = x + (size_t)s * DIN;      // batch row stride S*D

        // gates, x half: x_t @ [Wi|Wf|Wog|Wc]  (K tiles 0..15, f32 -> bf16)
#pragma unroll 4
        for (int kk = 0; kk < 16; ++kk) {
            v16bf a  = a_frag_f32(xbase, (long)SEQ * DIN, m0, kk << 5, lane);
            v16bf b0 = b_frag(Bp0, 128, kk, g,      lane);   // i
            v16bf b1 = b_frag(Bp0, 128, kk, 32 + g, lane);   // f
            v16bf b2 = b_frag(Bp0, 128, kk, 64 + g, lane);   // o
            v16bf b3 = b_frag(Bp0, 128, kk, 96 + g, lane);   // c~
            acc_i = WMMA_BF16(a, b0, acc_i);
            acc_f = WMMA_BF16(a, b1, acc_f);
            acc_o = WMMA_BF16(a, b2, acc_o);
            acc_c = WMMA_BF16(a, b3, acc_c);
        }
        // gates, h half: h @ [Ui|Uf|Uog|Uc]  (K tiles 16..31, bf16 state)
#pragma unroll 4
        for (int kk = 0; kk < 16; ++kk) {
            v16bf a  = a_frag_bf16(hsrc, HID, m0, kk << 5, lane);
            v16bf b0 = b_frag(Bp0, 128, kk + 16, g,      lane);
            v16bf b1 = b_frag(Bp0, 128, kk + 16, 32 + g, lane);
            v16bf b2 = b_frag(Bp0, 128, kk + 16, 64 + g, lane);
            v16bf b3 = b_frag(Bp0, 128, kk + 16, 96 + g, lane);
            acc_i = WMMA_BF16(a, b0, acc_i);
            acc_f = WMMA_BF16(a, b1, acc_f);
            acc_o = WMMA_BF16(a, b2, acc_o);
            acc_c = WMMA_BF16(a, b3, acc_c);
        }
        // decomposition: c @ W_decomp  (K = 512)
#pragma unroll 4
        for (int kk = 0; kk < 16; ++kk) {
            v16bf a = a_frag_bf16(csrc, HID, m0, kk << 5, lane);
            v16bf b = b_frag(Bp1, 32, kk, g, lane);
            acc_d = WMMA_BF16(a, b, acc_d);
        }

        // prefetch next step's x_t rows (the only HBM-streamed operand)
        if (s + 1 < SEQ) {
            const float* xn = x + (size_t)(s + 1) * DIN
                              + (long)(m0 + r) * (SEQ * DIN) + hf * 256;
            __builtin_prefetch(xn, 0, 1);
        }

        // gate math in registers: lane owns (row0+v, col) of every accumulator
#pragma unroll
        for (int v = 0; v < 8; ++v) {
            int row = row0 + v;
            size_t o = (size_t)row * HID + col;
            float t = time[(size_t)row * SEQ + s];
            float T = 1.0f / logf(t + ECONST);
            float cst = tanhf(acc_d[v] + Bd);
            float c = c32[o];
            c = c - cst + T * cst;
            float ig = 1.f / (1.f + __expf(-(acc_i[v] + Bi)));
            float fg = 1.f / (1.f + __expf(-(acc_f[v] + Bf)));
            float og = 1.f / (1.f + __expf(-(acc_o[v] + Bo)));
            float cn = tanhf(acc_c[v] + Bc);
            c = fg * c + ig * cn;
            float h = og * tanhf(c);
            c32[o]  = c;
            cdst[o] = (__bf16)c;
            hdst[o] = (__bf16)h;
            if (s == SEQ - 1) h32[o] = h;
        }

        grid_sync(barcnt, bargen);      // publish state for next step

        __bf16* tp;
        tp = hsrc; hsrc = hdst; hdst = tp;
        tp = csrc; csrc = cdst; cdst = tp;
    }
}

// ---------------------------------------------------------------------------
// Output head: out = relu(h @ Wo + bo) @ Wsm + bsm   ([256,2], negligible)
// ---------------------------------------------------------------------------
__global__ __launch_bounds__(64) void head_kernel(
    const float* __restrict__ h32, const float* __restrict__ Wo,
    const float* __restrict__ bo, const float* __restrict__ Wsm,
    const float* __restrict__ bsm, float* __restrict__ out) {
    __shared__ float fc[64];
    int b = blockIdx.x, j = threadIdx.x;
    float acc = bo[j];
    for (int k = 0; k < HID; ++k)
        acc += h32[(size_t)b * HID + k] * Wo[(size_t)k * 64 + j];
    fc[j] = acc > 0.f ? acc : 0.f;
    __syncthreads();
    if (j < 2) {
        float o = bsm[j];
        for (int k = 0; k < 64; ++k) o += fc[k] * Wsm[k * 2 + j];
        out[b * 2 + j] = o;
    }
}

// ---------------------------------------------------------------------------
// Host launcher
// ---------------------------------------------------------------------------
extern "C" void kernel_launch(void* const* d_in, const int* in_sizes, int n_in,
                              void* d_out, int out_size, void* d_ws, size_t ws_size,
                              hipStream_t stream) {
    (void)in_sizes; (void)n_in; (void)out_size; (void)ws_size;
    const float* x    = (const float*)d_in[0];
    const float* time = (const float*)d_in[1];
    const float* Wi   = (const float*)d_in[2];
    const float* Ui   = (const float*)d_in[3];
    const float* bi   = (const float*)d_in[4];
    const float* Wf   = (const float*)d_in[5];
    const float* Uf   = (const float*)d_in[6];
    const float* bfv  = (const float*)d_in[7];
    const float* Wog  = (const float*)d_in[8];
    const float* Uog  = (const float*)d_in[9];
    const float* bog  = (const float*)d_in[10];
    const float* Wc   = (const float*)d_in[11];
    const float* Uc   = (const float*)d_in[12];
    const float* bc   = (const float*)d_in[13];
    const float* Wd   = (const float*)d_in[14];
    const float* bd   = (const float*)d_in[15];
    const float* Wo   = (const float*)d_in[16];
    const float* bo   = (const float*)d_in[17];
    const float* Wsm  = (const float*)d_in[18];
    const float* bsm  = (const float*)d_in[19];
    float* out = (float*)d_out;

    // workspace carve-out (~6.6 MB)
    char* w = (char*)d_ws;
    __bf16* Bp0  = (__bf16*)w; w += (size_t)1024 * 2048 * 2;   // 4 MB
    __bf16* Bp1  = (__bf16*)w; w += (size_t)512 * 512 * 2;     // 512 KB
    float*  c32  = (float*) w; w += (size_t)256 * 512 * 4;     // 512 KB
    float*  h32  = (float*) w; w += (size_t)256 * 512 * 4;     // 512 KB
    __bf16* hbf0 = (__bf16*)w; w += (size_t)256 * 512 * 2;     // 256 KB
    __bf16* cbf0 = (__bf16*)w; w += (size_t)256 * 512 * 2;
    __bf16* hbf1 = (__bf16*)w; w += (size_t)256 * 512 * 2;
    __bf16* cbf1 = (__bf16*)w; w += (size_t)256 * 512 * 2;
    unsigned* barcnt = (unsigned*)w; w += 256;                 // padded
    unsigned* bargen = (unsigned*)w; w += 256;

    // 1) pack weights into WMMA B-fragment layout
    {
        int tiles = 32 * 128 + 16 * 32;
        int threads = tiles * 32;
        pack_weights_kernel<<<(threads + 255) / 256, 256, 0, stream>>>(
            Wi, Wf, Wog, Wc, Ui, Uf, Uog, Uc, Wd, Bp0, Bp1);
    }
    // 2) reset device-wide barrier
    zero_barrier_kernel<<<1, 1, 0, stream>>>(barcnt, bargen);

    // 3) whole recurrence in ONE persistent kernel (512 steps internally)
    tlstm_persistent_kernel<<<NBLK, 256, 0, stream>>>(
        x, time, Bp0, Bp1, bi, bfv, bog, bc, bd,
        c32, h32, hbf0, cbf0, hbf1, cbf1, barcnt, bargen);

    // 4) output head
    head_kernel<<<256, 64, 0, stream>>>(h32, Wo, bo, Wsm, bsm, out);
}
